// GCN_Net_48206712930319
// MI455X (gfx1250) — matrix operand order repaired and checked by reading, not compiled, not run on hardware
//
#include <hip/hip_runtime.h>
#include <hip/hip_bf16.h>

// Problem constants (match reference)
#define N_NODES 100000
#define N_EDGES 1600000
#define F_IN    512
#define HID     256
#define N_CLS   40

typedef __attribute__((ext_vector_type(16))) __bf16         v16bf;
typedef __attribute__((ext_vector_type(16))) unsigned short v16us;
typedef __attribute__((ext_vector_type(8)))  float          v8f;
typedef __attribute__((ext_vector_type(4)))  float          v4f;

union BF16Frag { v16us u; v16bf b; };

__device__ __forceinline__ unsigned short f32_to_bf16_rne(float f) {
  unsigned int u = __float_as_uint(f);
  u += 0x7FFFu + ((u >> 16) & 1u);
  return (unsigned short)(u >> 16);
}

__device__ __forceinline__ float atomic_add_f32_agent(float* p, float v) {
  return __hip_atomic_fetch_add(p, v, __ATOMIC_RELAXED, __HIP_MEMORY_SCOPE_AGENT);
}

// ---------------------------------------------------------------------------
// Degree / dinv
// ---------------------------------------------------------------------------
__global__ void deg_init_kernel(float* __restrict__ deg) {
  int i = blockIdx.x * blockDim.x + threadIdx.x;
  if (i < N_NODES) deg[i] = 1.0f;  // self-loop
}

__global__ void deg_count_kernel(const int* __restrict__ dst, float* __restrict__ deg) {
  int e = blockIdx.x * blockDim.x + threadIdx.x;
  if (e < N_EDGES) {
    int d = dst[e];
    atomic_add_f32_agent(&deg[d], 1.0f);
  }
}

__global__ void deg_rsqrt_kernel(float* __restrict__ deg) {
  int i = blockIdx.x * blockDim.x + threadIdx.x;
  if (i < N_NODES) deg[i] = rsqrtf(deg[i]);
}

// ---------------------------------------------------------------------------
// Pre-pack B = W[KDIM][NDIM] (f32) into bf16 WMMA B-fragments, padded to 16
// columns per tile. Fragment layout (wave32, CDNA5 ISA):
//   lane -> col = tile*16 + (lane&15); element e <-> K = kk32*32 + (lane>>4)*8 + (e&7) + ((e&8)<<1)
// Storage: Bpack[((tile*(KDIM/32) + kk32)*32 + lane)*16 + e]
// One thread per (tile, kk32, lane): writes 16 contiguous bf16 (32 bytes).
// ---------------------------------------------------------------------------
template <int KDIM, int NDIM, int NTILES>
__global__ void pack_b_kernel(const float* __restrict__ W, unsigned short* __restrict__ Bpack) {
  int idx = blockIdx.x * blockDim.x + threadIdx.x;     // (tile, kk32, lane)
  constexpr int KCH = KDIM / 32;
  if (idx >= NTILES * KCH * 32) return;
  int lane = idx & 31;
  int kk32 = (idx >> 5) % KCH;
  int tile = (idx >> 5) / KCH;
  int col  = tile * 16 + (lane & 15);
  int kbase = kk32 * 32 + (lane >> 4) * 8;
  v16us out;
#pragma unroll
  for (int e = 0; e < 16; ++e) {
    float v = (col < NDIM) ? W[(size_t)(kbase + (e & 7) + ((e & 8) << 1)) * NDIM + col] : 0.0f;
    out[e] = f32_to_bf16_rne(v);
  }
  *(v16us*)(Bpack + (size_t)idx * 16) = out;
}

// ---------------------------------------------------------------------------
// WMMA GEMM: C[M,NDIM] = A[M,KDIM] (fp32, converted to bf16 on the fly) x
// Bpack (pre-packed bf16 fragments), f32 accumulate.
// One wave computes a 16 x (NT*16) strip: the A fragment is loaded/converted
// once per K-step and reused by NT WMMAs -> A streamed n_tiles/NT times total.
//  - GEMM1: NT=8 -> X read twice (~410 MB HBM vs 3.3 GB for NT=1)
//  - GEMM2: NT=3 -> x1 read once
// A fragment: lane -> row = lane&15, half = lane>>4;
//   element e <-> K = half*8 + (e&7) + ((e&8)<<1) (two contiguous 8-float runs)
// C/D: VGPR r <-> M = r + half*8, N = lane&15.
// ---------------------------------------------------------------------------
template <int KDIM, int NDIM, int NT>
__global__ __launch_bounds__(256)
void wmma_gemm_strip_kernel(const float* __restrict__ A,
                            const unsigned short* __restrict__ Bpack,
                            float* __restrict__ C, int n_groups, int total_waves) {
  constexpr int KCH = KDIM / 32;
  int wave = (int)((blockIdx.x * blockDim.x + threadIdx.x) >> 5);
  if (wave >= total_waves) return;
  int lane    = threadIdx.x & 31;
  int m_tile  = wave / n_groups;
  int group   = wave - m_tile * n_groups;
  int n_tile0 = group * NT;
  int half    = lane >> 4;
  int sub     = lane & 15;

  const float* Arow = A + (size_t)(m_tile * 16 + sub) * KDIM;
  // Per-lane base into packed B for this wave's first tile.
  const unsigned short* Bbase =
      Bpack + ((size_t)n_tile0 * KCH * 32 + lane) * 16;

  v8f acc[NT];
#pragma unroll
  for (int t = 0; t < NT; ++t) acc[t] = (v8f){};

#pragma unroll 2
  for (int kk32 = 0; kk32 < KCH; ++kk32) {
    // ---- A fragment: two contiguous 8-float runs per lane, convert RNE.
    const float* p0 = Arow + kk32 * 32 + half * 8;
    __builtin_prefetch(p0 + 64, 0, 1);  // stream next A chunk (global_prefetch_b8)
    v4f a0 = *(const v4f*)(p0);
    v4f a1 = *(const v4f*)(p0 + 4);
    v4f a2 = *(const v4f*)(p0 + 16);
    v4f a3 = *(const v4f*)(p0 + 20);
    BF16Frag a;
#pragma unroll
    for (int j = 0; j < 4; ++j) {
      a.u[j]      = f32_to_bf16_rne(a0[j]);
      a.u[j + 4]  = f32_to_bf16_rne(a1[j]);
      a.u[j + 8]  = f32_to_bf16_rne(a2[j]);
      a.u[j + 12] = f32_to_bf16_rne(a3[j]);
    }
    // ---- NT WMMAs sharing this A fragment; B fragments are contiguous 32 B.
#pragma unroll
    for (int t = 0; t < NT; ++t) {
      BF16Frag bfr;
      bfr.u = *(const v16us*)(Bbase + ((size_t)t * KCH + kk32) * 32 * 16);
      acc[t] = __builtin_amdgcn_wmma_f32_16x16x32_bf16(
          /*neg_a=*/false, a.b, /*neg_b=*/false, bfr.b,
          /*c_mod=*/(short)0, acc[t], /*reuse_a=*/false, /*reuse_b=*/false);
    }
  }

#pragma unroll
  for (int t = 0; t < NT; ++t) {
    int ncol = (n_tile0 + t) * 16 + sub;
    if ((NDIM % 16 == 0) || ncol < NDIM) {
#pragma unroll
      for (int r = 0; r < 8; ++r) {
        int m = m_tile * 16 + r + half * 8;  // C/D layout: VGPR r -> M = r + half*8
        C[(size_t)m * NDIM + ncol] = acc[t][r];
      }
    }
  }
}

// ---------------------------------------------------------------------------
// agg[i][f] = h[i][f] * dinv[i]^2  (self-loop term, also initializes agg)
// ---------------------------------------------------------------------------
__global__ void self_loop_init_kernel(const float* __restrict__ h,
                                      const float* __restrict__ dinv,
                                      float* __restrict__ agg, int F) {
  int node = blockIdx.x;
  float di = dinv[node];
  float s  = di * di;
  for (int f = threadIdx.x; f < F; f += blockDim.x) {
    size_t idx = (size_t)node * F + f;
    agg[idx] = h[idx] * s;
  }
}

// ---------------------------------------------------------------------------
// Edge scatter, F = 256: one block per edge, one lane per feature.
// h1 + agg1 (~205 MB) live mostly in the 192 MB L2 -> L2-bandwidth bound,
// native fp32 global atomics for the scatter.
// ---------------------------------------------------------------------------
__global__ __launch_bounds__(HID)
void edge_scatter_hid_kernel(const int* __restrict__ src, const int* __restrict__ dst,
                             const float* __restrict__ dinv, const float* __restrict__ h,
                             float* __restrict__ agg) {
  int e = blockIdx.x;
  int s = src[e];
  int d = dst[e];
  float norm = dinv[s] * dinv[d];
  int f = threadIdx.x;
  float v = h[(size_t)s * HID + f] * norm;
  atomic_add_f32_agent(&agg[(size_t)d * HID + f], v);
}

// ---------------------------------------------------------------------------
// Edge scatter, F = 40: 8 edges per 320-thread block.
// ---------------------------------------------------------------------------
__global__ __launch_bounds__(320)
void edge_scatter_cls_kernel(const int* __restrict__ src, const int* __restrict__ dst,
                             const float* __restrict__ dinv, const float* __restrict__ h,
                             float* __restrict__ agg) {
  int t = threadIdx.x;               // 0..319
  int e = blockIdx.x * 8 + t / N_CLS;
  int f = t % N_CLS;
  if (e >= N_EDGES) return;
  int s = src[e];
  int d = dst[e];
  float norm = dinv[s] * dinv[d];
  float v = h[(size_t)s * N_CLS + f] * norm;
  atomic_add_f32_agent(&agg[(size_t)d * N_CLS + f], v);
}

// ---------------------------------------------------------------------------
// x = relu(x + b), in place, F = 256
// ---------------------------------------------------------------------------
__global__ __launch_bounds__(HID)
void bias_relu_kernel(float* __restrict__ x, const float* __restrict__ b) {
  int node = blockIdx.x;
  int f = threadIdx.x;
  size_t idx = (size_t)node * HID + f;
  x[idx] = fmaxf(x[idx] + b[f], 0.0f);
}

// ---------------------------------------------------------------------------
// Final: logits z = agg2 + b2; write log_softmax and softmax.
// One wave (32 lanes) per node; lane handles f = lane and f = lane+32 (<40).
// ---------------------------------------------------------------------------
__global__ __launch_bounds__(256)
void softmax_out_kernel(const float* __restrict__ agg2, const float* __restrict__ b2,
                        float* __restrict__ out_ls, float* __restrict__ out_sm) {
  int node = (int)((blockIdx.x * blockDim.x + threadIdx.x) >> 5);
  int lane = threadIdx.x & 31;
  if (node >= N_NODES) return;
  const float* row = agg2 + (size_t)node * N_CLS;

  float z0 = (lane < N_CLS) ? row[lane] + b2[lane] : -__builtin_inff();
  float z1 = (lane + 32 < N_CLS) ? row[lane + 32] + b2[lane + 32] : -__builtin_inff();

  float m = fmaxf(z0, z1);
#pragma unroll
  for (int off = 16; off > 0; off >>= 1) m = fmaxf(m, __shfl_xor(m, off, 32));

  float e0 = (lane < N_CLS) ? expf(z0 - m) : 0.0f;
  float e1 = (lane + 32 < N_CLS) ? expf(z1 - m) : 0.0f;
  float s = e0 + e1;
#pragma unroll
  for (int off = 16; off > 0; off >>= 1) s += __shfl_xor(s, off, 32);

  float ls  = logf(s);
  float inv = 1.0f / s;
  if (lane < N_CLS) {
    out_ls[(size_t)node * N_CLS + lane] = z0 - m - ls;
    out_sm[(size_t)node * N_CLS + lane] = e0 * inv;
  }
  if (lane + 32 < N_CLS) {
    out_ls[(size_t)node * N_CLS + lane + 32] = z1 - m - ls;
    out_sm[(size_t)node * N_CLS + lane + 32] = e1 * inv;
  }
}

// ---------------------------------------------------------------------------
// Launch
// ---------------------------------------------------------------------------
extern "C" void kernel_launch(void* const* d_in, const int* in_sizes, int n_in,
                              void* d_out, int out_size, void* d_ws, size_t ws_size,
                              hipStream_t stream) {
  const float* X  = (const float*)d_in[0];   // [100000, 512]
  const float* W1 = (const float*)d_in[1];   // [512, 256]
  const float* b1 = (const float*)d_in[2];   // [256]
  const float* W2 = (const float*)d_in[3];   // [256, 40]
  const float* b2 = (const float*)d_in[4];   // [40]
  const int*   ei = (const int*)d_in[5];     // [2, 1600000] int32
  const int* src = ei;
  const int* dst = ei + N_EDGES;

  // Workspace carve-up (~238 MB total):
  //   dinv | h1 (100000x256) | x1/agg1 (100000x256) | h2 (100000x40)
  //   agg2 (100000x40) | Bpack1 (16*16*512 bf16) | Bpack2 (3*8*512 bf16)
  char* w = (char*)d_ws;
  float* dinv = (float*)w;  w += (((size_t)N_NODES * 4) + 255) & ~(size_t)255;
  float* h1   = (float*)w;  w += (size_t)N_NODES * HID * 4;
  float* x1   = (float*)w;  w += (size_t)N_NODES * HID * 4;       // agg1, then relu'd in place
  float* h2   = (float*)w;  w += (((size_t)N_NODES * N_CLS * 4) + 255) & ~(size_t)255;
  float* agg2 = (float*)w;  w += (((size_t)N_NODES * N_CLS * 4) + 255) & ~(size_t)255;
  unsigned short* Bpack1 = (unsigned short*)w;  w += (size_t)16 * (F_IN / 32) * 512 * 2;
  unsigned short* Bpack2 = (unsigned short*)w;

  float* out_ls = (float*)d_out;                       // [100000, 40]
  float* out_sm = out_ls + (size_t)N_NODES * N_CLS;    // [100000, 40]

  // 1) degrees -> dinv ; pack weights into WMMA B-fragment layout (bf16)
  deg_init_kernel<<<(N_NODES + 255) / 256, 256, 0, stream>>>(dinv);
  deg_count_kernel<<<(N_EDGES + 255) / 256, 256, 0, stream>>>(dst, dinv);
  deg_rsqrt_kernel<<<(N_NODES + 255) / 256, 256, 0, stream>>>(dinv);
  pack_b_kernel<F_IN, HID, 16><<<(16 * (F_IN / 32) * 32 + 255) / 256, 256, 0, stream>>>(W1, Bpack1);
  pack_b_kernel<HID, N_CLS, 3><<<(3 * (HID / 32) * 32 + 255) / 256, 256, 0, stream>>>(W2, Bpack2);

  // 2) h1 = X @ W1   (M=100000, K=512, N=256): 16 n-tiles, NT=8 -> 2 groups
  {
    int n_groups = 2;
    int total = (N_NODES / 16) * n_groups;          // 12500 waves
    wmma_gemm_strip_kernel<F_IN, HID, 8>
        <<<(total + 7) / 8, 256, 0, stream>>>(X, Bpack1, h1, n_groups, total);
  }

  // 3) agg1 = h1 * dinv^2 ; scatter edges ; x1 = relu(agg1 + b1)
  self_loop_init_kernel<<<N_NODES, 256, 0, stream>>>(h1, dinv, x1, HID);
  edge_scatter_hid_kernel<<<N_EDGES, HID, 0, stream>>>(src, dst, dinv, h1, x1);
  bias_relu_kernel<<<N_NODES, HID, 0, stream>>>(x1, b1);

  // 4) h2 = x1 @ W2  (M=100000, K=256, N=40 -> 3 padded tiles, NT=3, 1 group)
  {
    int n_groups = 1;
    int total = (N_NODES / 16);                     // 6250 waves
    wmma_gemm_strip_kernel<HID, N_CLS, 3>
        <<<(total + 7) / 8, 256, 0, stream>>>(x1, Bpack2, h2, n_groups, total);
  }

  // 5) agg2 = h2 * dinv^2 ; scatter edges
  self_loop_init_kernel<<<N_NODES, 64, 0, stream>>>(h2, dinv, agg2, N_CLS);
  edge_scatter_cls_kernel<<<(N_EDGES + 7) / 8, 320, 0, stream>>>(src, dst, dinv, h2, agg2);

  // 6) z = agg2 + b2 ; emit log_softmax and softmax (one wave per node)
  softmax_out_kernel<<<(N_NODES * 32 + 255) / 256, 256, 0, stream>>>(agg2, b2, out_ls, out_sm);
}
